// MoE_58875411694139
// MI455X (gfx1250) — compile-verified, matrix operand order checked by reading, not run on hardware
//
#include <hip/hip_runtime.h>

// ---------------- problem constants (from setup_inputs) ----------------
#define NTOK   16384      // B*S
#define DMODEL 1024
#define FFN    2048
#define NEXP   8
#define TOPK   2
#define CAP    2560
#define NK     (NTOK*TOPK)      // 32768
#define NSLOTS (NEXP*CAP)       // 20480

typedef __bf16 bf16;
typedef __attribute__((ext_vector_type(16))) __bf16 v16bf;
typedef __attribute__((ext_vector_type(8)))  __bf16 v8bf;
typedef __attribute__((ext_vector_type(4)))  __bf16 v4bf;
typedef __attribute__((ext_vector_type(8)))  float  v8f;
// exact parameter type of __builtin_amdgcn_global_load_async_to_lds_b128
// (from hipcc diagnostic: "__attribute__((__vector_size__(4 * sizeof(int)))) int *")
typedef int moe_v4i __attribute__((vector_size(4 * sizeof(int))));

// ---- optional CDNA5 async global->LDS path (guarded; falls back cleanly) ----
#if __has_builtin(__builtin_amdgcn_global_load_async_to_lds_b128)
#define MOE_ASYNC_LDS 1
#else
#define MOE_ASYNC_LDS 0
#endif

__device__ __forceinline__ void moe_wait_async() {
#if MOE_ASYNC_LDS
#if __has_builtin(__builtin_amdgcn_s_wait_asynccnt)
    __builtin_amdgcn_s_wait_asynccnt(0);
#else
    asm volatile("s_wait_asynccnt 0" ::: "memory");
#endif
#endif
}

__device__ __forceinline__ void moe_copy16x2_to_lds(const uint4* src, uint4* dst) {
#if MOE_ASYNC_LDS
    __builtin_amdgcn_global_load_async_to_lds_b128(
        (moe_v4i*)src, (moe_v4i*)dst, 0, 0);
    __builtin_amdgcn_global_load_async_to_lds_b128(
        (moe_v4i*)(src + 1), (moe_v4i*)(dst + 1), 0, 0);
#else
    dst[0] = src[0];
    dst[1] = src[1];
#endif
}

// ---------------- zero fill ----------------
__global__ void moe_zero_u64(unsigned long long* __restrict__ p, long long n) {
    long long i = (long long)blockIdx.x * blockDim.x + threadIdx.x;
    long long stride = (long long)gridDim.x * blockDim.x;
    for (; i < n; i += stride) p[i] = 0ull;
}

// ---------------- router: logits -> softmax -> top2 ----------------
__global__ __launch_bounds__(256) void moe_router(const float* __restrict__ x,
                                                  const float* __restrict__ gw,
                                                  float* __restrict__ probs,
                                                  float* __restrict__ wts,
                                                  int* __restrict__ idx) {
    const int lane = threadIdx.x & 31;
    const int wv   = threadIdx.x >> 5;
    const int token = blockIdx.x * 8 + wv;
    const float* xr = x + (size_t)token * DMODEL;

    float acc[8];
    #pragma unroll
    for (int e = 0; e < 8; ++e) acc[e] = 0.f;

    for (int d = lane; d < DMODEL; d += 32) {
        float xv = xr[d];
        const float* g = gw + (size_t)d * 8;
        float4 g0 = *(const float4*)g;
        float4 g1 = *(const float4*)(g + 4);
        acc[0] += xv * g0.x; acc[1] += xv * g0.y; acc[2] += xv * g0.z; acc[3] += xv * g0.w;
        acc[4] += xv * g1.x; acc[5] += xv * g1.y; acc[6] += xv * g1.z; acc[7] += xv * g1.w;
    }
    #pragma unroll
    for (int off = 16; off > 0; off >>= 1)
        #pragma unroll
        for (int e = 0; e < 8; ++e) acc[e] += __shfl_xor(acc[e], off, 32);

    if (lane == 0) {
        float m = acc[0];
        #pragma unroll
        for (int e = 1; e < 8; ++e) m = fmaxf(m, acc[e]);
        float p[8]; float s = 0.f;
        #pragma unroll
        for (int e = 0; e < 8; ++e) { p[e] = __expf(acc[e] - m); s += p[e]; }
        float inv = 1.f / s;
        #pragma unroll
        for (int e = 0; e < 8; ++e) { p[e] *= inv; probs[(size_t)token * 8 + e] = p[e]; }
        int i0 = 0; float v0 = p[0];
        #pragma unroll
        for (int e = 1; e < 8; ++e) if (p[e] > v0) { v0 = p[e]; i0 = e; }
        int i1 = -1; float v1 = -1.f;
        #pragma unroll
        for (int e = 0; e < 8; ++e) if (e != i0 && p[e] > v1) { v1 = p[e]; i1 = e; }
        float wsum = v0 + v1;
        wts[token * 2 + 0] = v0 / wsum;
        wts[token * 2 + 1] = v1 / wsum;
        idx[token * 2 + 0] = i0;
        idx[token * 2 + 1] = i1;
    }
}

// ---------------- deterministic capacity scan (1 workgroup) ----------------
__global__ __launch_bounds__(1024) void moe_scan(const int* __restrict__ idxF,
                                                 int* __restrict__ slot) {
    __shared__ unsigned sc[1024 * 8];
    const int t = threadIdx.x;
    const int r0 = t * 32;                 // 32 rows per thread, 1024*32 = 32768
    int myE[32];
    int hist[8];
    #pragma unroll
    for (int j = 0; j < 8; ++j) hist[j] = 0;
    for (int i = 0; i < 32; ++i) {
        int e = idxF[r0 + i];
        myE[i] = e;
        #pragma unroll
        for (int j = 0; j < 8; ++j) hist[j] += (e == j);
    }
    #pragma unroll
    for (int j = 0; j < 8; ++j) sc[t * 8 + j] = (unsigned)hist[j];
    __syncthreads();
    for (int off = 1; off < 1024; off <<= 1) {
        unsigned v[8];
        #pragma unroll
        for (int j = 0; j < 8; ++j) v[j] = (t >= off) ? sc[(t - off) * 8 + j] : 0u;
        __syncthreads();
        if (t >= off) {
            #pragma unroll
            for (int j = 0; j < 8; ++j) sc[t * 8 + j] += v[j];
        }
        __syncthreads();
    }
    int base[8], run[8];
    #pragma unroll
    for (int j = 0; j < 8; ++j) { base[j] = (int)sc[t * 8 + j] - hist[j]; run[j] = 0; }
    for (int i = 0; i < 32; ++i) {
        int e = myE[i];
        int rank = 0;
        #pragma unroll
        for (int j = 0; j < 8; ++j) { run[j] += (e == j); rank += (e == j) ? (base[j] + run[j]) : 0; }
        slot[r0 + i] = (rank <= CAP) ? (e * CAP + rank - 1) : -1;
    }
}

// ---------------- dispatch scatter f32 -> bf16 ----------------
__global__ __launch_bounds__(256) void moe_dispatch(const float* __restrict__ x,
                                                    const int* __restrict__ slot,
                                                    bf16* __restrict__ disp) {
    const int r = blockIdx.x;
    const int s = slot[r];
    if (s < 0) return;
    const int token = r >> 1;
    const float4 v = ((const float4*)(x + (size_t)token * DMODEL))[threadIdx.x];
    v4bf o; o.x = (bf16)v.x; o.y = (bf16)v.y; o.z = (bf16)v.z; o.w = (bf16)v.w;
    *(v4bf*)(disp + (size_t)s * DMODEL + threadIdx.x * 4) = o;
}

// ---------------- expert GEMM 1: gate/up + SwiGLU (bf16 WMMA) ----------------
// A: disp[e] [CAP x DMODEL] bf16 row-major. Bg/Bu: f32 [DMODEL x FFN] row-major.
// H: [CAP x FFN] bf16.
__global__ __launch_bounds__(256) void moe_gateup(const bf16* __restrict__ A,
                                                  const float* __restrict__ Bg,
                                                  const float* __restrict__ Bu,
                                                  bf16* __restrict__ H) {
    __shared__ bf16 As[128][32];
    __shared__ bf16 Bgs[128][32];   // transposed: [n][k]
    __shared__ bf16 Bus[128][32];
    const int tid = threadIdx.x, lane = tid & 31, wv = tid >> 5;
    const int mW = (wv & 1) * 64, nW = (wv >> 1) * 32;
    const int cBase = blockIdx.y * 128, nBase = blockIdx.x * 128;
    const int rA = tid >> 1, hA = tid & 1;
    // B staging: thread owns an n-pair (2*np, 2*np+1) within an 8-row K-quarter.
    const int np = tid & 63, kq = tid >> 6;

    v8f accG[4][2] = {};
    v8f accU[4][2] = {};

    for (int kt = 0; kt < DMODEL; kt += 32) {
        // stage A (already bf16): 32B per thread (async-to-LDS if available)
        const uint4* srcA = (const uint4*)(A + (size_t)(cBase + rA) * DMODEL + kt + hA * 16);
        uint4* dstA = (uint4*)&As[rA][hA * 16];
        moe_copy16x2_to_lds(srcA, dstA);

        // stage B tiles: coalesced float2 column reads, f32->bf16, two b128 LDS stores
        const float* pg = Bg + (size_t)(kt + kq * 8) * FFN + nBase + 2 * np;
        const float* pu = Bu + (size_t)(kt + kq * 8) * FFN + nBase + 2 * np;
        if (kt + 32 < DMODEL) {
            __builtin_prefetch(pg + (size_t)32 * FFN, 0, 1);
            __builtin_prefetch(pu + (size_t)32 * FFN, 0, 1);
        }
        v8bf g0, g1, u0, u1;
        #pragma unroll
        for (int j = 0; j < 8; ++j) {
            float2 vg = *(const float2*)(pg + (size_t)j * FFN);
            float2 vu = *(const float2*)(pu + (size_t)j * FFN);
            g0[j] = (bf16)vg.x; g1[j] = (bf16)vg.y;
            u0[j] = (bf16)vu.x; u1[j] = (bf16)vu.y;
        }
        *(v8bf*)&Bgs[2 * np + 0][kq * 8] = g0;
        *(v8bf*)&Bgs[2 * np + 1][kq * 8] = g1;
        *(v8bf*)&Bus[2 * np + 0][kq * 8] = u0;
        *(v8bf*)&Bus[2 * np + 1][kq * 8] = u1;

        moe_wait_async();
        __syncthreads();

        const int fr = lane & 15, kh = (lane >> 4) * 16;
        v16bf aF[4], gF[2], uF[2];
        #pragma unroll
        for (int mi = 0; mi < 4; ++mi) aF[mi] = *(const v16bf*)&As[mW + mi * 16 + fr][kh];
        #pragma unroll
        for (int ni = 0; ni < 2; ++ni) {
            gF[ni] = *(const v16bf*)&Bgs[nW + ni * 16 + fr][kh];
            uF[ni] = *(const v16bf*)&Bus[nW + ni * 16 + fr][kh];
        }
        #pragma unroll
        for (int mi = 0; mi < 4; ++mi)
            #pragma unroll
            for (int ni = 0; ni < 2; ++ni) {
                accG[mi][ni] = __builtin_amdgcn_wmma_f32_16x16x32_bf16(
                    false, aF[mi], false, gF[ni], (short)0, accG[mi][ni], false, false);
                accU[mi][ni] = __builtin_amdgcn_wmma_f32_16x16x32_bf16(
                    false, aF[mi], false, uF[ni], (short)0, accU[mi][ni], false, false);
            }
        __syncthreads();
    }
    // epilogue: SwiGLU, store bf16 hidden
    const int hi = (lane >> 4) * 8, cn0 = lane & 15;
    #pragma unroll
    for (int mi = 0; mi < 4; ++mi)
        #pragma unroll
        for (int ni = 0; ni < 2; ++ni)
            #pragma unroll
            for (int v = 0; v < 8; ++v) {
                int rm = cBase + mW + mi * 16 + hi + v;
                int cn = nBase + nW + ni * 16 + cn0;
                float g = accG[mi][ni][v], u = accU[mi][ni][v];
                float h = (g / (1.0f + __expf(-g))) * u;
                H[(size_t)rm * FFN + cn] = (bf16)h;
            }
}

// ---------------- expert GEMM 2: down projection ----------------
// A: H [CAP x FFN] bf16. Bd: f32 [FFN x DMODEL]. O: f32 [CAP x DMODEL] (expert slice).
__global__ __launch_bounds__(256) void moe_down(const bf16* __restrict__ A,
                                                const float* __restrict__ Bd,
                                                float* __restrict__ O) {
    __shared__ bf16 As[128][32];
    __shared__ bf16 Bs[128][32];   // transposed [n][k]
    const int tid = threadIdx.x, lane = tid & 31, wv = tid >> 5;
    const int mW = (wv & 1) * 64, nW = (wv >> 1) * 32;
    const int cBase = blockIdx.y * 128, nBase = blockIdx.x * 128;
    const int rA = tid >> 1, hA = tid & 1;
    const int np = tid & 63, kq = tid >> 6;

    v8f acc[4][2] = {};

    for (int kt = 0; kt < FFN; kt += 32) {
        const uint4* srcA = (const uint4*)(A + (size_t)(cBase + rA) * FFN + kt + hA * 16);
        uint4* dstA = (uint4*)&As[rA][hA * 16];
        moe_copy16x2_to_lds(srcA, dstA);

        const float* pb = Bd + (size_t)(kt + kq * 8) * DMODEL + nBase + 2 * np;
        if (kt + 32 < FFN) __builtin_prefetch(pb + (size_t)32 * DMODEL, 0, 1);
        v8bf b0, b1;
        #pragma unroll
        for (int j = 0; j < 8; ++j) {
            float2 vb = *(const float2*)(pb + (size_t)j * DMODEL);
            b0[j] = (bf16)vb.x; b1[j] = (bf16)vb.y;
        }
        *(v8bf*)&Bs[2 * np + 0][kq * 8] = b0;
        *(v8bf*)&Bs[2 * np + 1][kq * 8] = b1;

        moe_wait_async();
        __syncthreads();

        const int fr = lane & 15, kh = (lane >> 4) * 16;
        v16bf aF[4], bF[2];
        #pragma unroll
        for (int mi = 0; mi < 4; ++mi) aF[mi] = *(const v16bf*)&As[mW + mi * 16 + fr][kh];
        #pragma unroll
        for (int ni = 0; ni < 2; ++ni) bF[ni] = *(const v16bf*)&Bs[nW + ni * 16 + fr][kh];
        #pragma unroll
        for (int mi = 0; mi < 4; ++mi)
            #pragma unroll
            for (int ni = 0; ni < 2; ++ni)
                acc[mi][ni] = __builtin_amdgcn_wmma_f32_16x16x32_bf16(
                    false, aF[mi], false, bF[ni], (short)0, acc[mi][ni], false, false);
        __syncthreads();
    }
    const int hi = (lane >> 4) * 8, cn0 = lane & 15;
    #pragma unroll
    for (int mi = 0; mi < 4; ++mi)
        #pragma unroll
        for (int ni = 0; ni < 2; ++ni)
            #pragma unroll
            for (int v = 0; v < 8; ++v) {
                int rm = cBase + mW + mi * 16 + hi + v;
                int cn = nBase + nW + ni * 16 + cn0;
                O[(size_t)rm * DMODEL + cn] = acc[mi][ni][v];
            }
}

// ---------------- combine ----------------
__global__ __launch_bounds__(256) void moe_combine(const float* __restrict__ eo,
                                                   const int* __restrict__ slot,
                                                   const float* __restrict__ wts,
                                                   float* __restrict__ out) {
    const int t = blockIdx.x, tid = threadIdx.x;
    const int s0 = slot[t * 2 + 0], s1 = slot[t * 2 + 1];
    const float w0 = wts[t * 2 + 0], w1 = wts[t * 2 + 1];
    float rx = 0.f, ry = 0.f, rz = 0.f, rw = 0.f;
    if (s0 >= 0) {
        float4 a = ((const float4*)(eo + (size_t)s0 * DMODEL))[tid];
        rx += w0 * a.x; ry += w0 * a.y; rz += w0 * a.z; rw += w0 * a.w;
    }
    if (s1 >= 0) {
        float4 a = ((const float4*)(eo + (size_t)s1 * DMODEL))[tid];
        rx += w1 * a.x; ry += w1 * a.y; rz += w1 * a.z; rw += w1 * a.w;
    }
    float4 r; r.x = rx; r.y = ry; r.z = rz; r.w = rw;
    ((float4*)(out + (size_t)t * DMODEL))[tid] = r;
}

// ---------------- aux loss (deterministic single block) ----------------
__global__ __launch_bounds__(256) void moe_aux(const int* __restrict__ idx,
                                               const float* __restrict__ probs,
                                               float* __restrict__ auxOut) {
    __shared__ float sF[8][256];
    __shared__ float sP[8][256];
    const int tid = threadIdx.x;
    float fc[8], pp[8];
    #pragma unroll
    for (int e = 0; e < 8; ++e) { fc[e] = 0.f; pp[e] = 0.f; }
    for (int t = tid; t < NTOK; t += 256) {
        int i0 = idx[t * 2 + 0], i1 = idx[t * 2 + 1];
        #pragma unroll
        for (int e = 0; e < 8; ++e) {
            fc[e] += (float)((i0 == e) + (i1 == e));
            pp[e] += probs[(size_t)t * 8 + e];
        }
    }
    #pragma unroll
    for (int e = 0; e < 8; ++e) { sF[e][tid] = fc[e]; sP[e][tid] = pp[e]; }
    __syncthreads();
    if (tid < 8) {
        float f = 0.f, p = 0.f;
        for (int i = 0; i < 256; ++i) { f += sF[tid][i]; p += sP[tid][i]; }
        sF[tid][0] = f; sP[tid][0] = p;
    }
    __syncthreads();
    if (tid == 0) {
        float s = 0.f;
        for (int e = 0; e < 8; ++e) {
            float f = sF[e][0] / ((float)NK + 1e-9f);
            float P = sP[e][0] / (float)NTOK;
            s += f * P;
        }
        auxOut[0] = (float)NEXP * s * 0.01f;
    }
}

// ---------------- launcher ----------------
extern "C" void kernel_launch(void* const* d_in, const int* in_sizes, int n_in,
                              void* d_out, int out_size, void* d_ws, size_t ws_size,
                              hipStream_t stream) {
    const float* x      = (const float*)d_in[0];
    const float* gate_w = (const float*)d_in[1];
    const float* w_gate = (const float*)d_in[2];
    const float* w_up   = (const float*)d_in[3];
    const float* w_down = (const float*)d_in[4];
    float* out = (float*)d_out;

    char* ws = (char*)d_ws;
    size_t off = 0;
    bf16* disp   = (bf16*)(ws + off);  off += (size_t)NSLOTS * DMODEL * sizeof(bf16);   // 40 MB
    float* eout  = (float*)(ws + off); off += (size_t)NSLOTS * DMODEL * sizeof(float);  // 80 MB
    bf16* hidden = (bf16*)(ws + off);  off += (size_t)CAP * FFN * sizeof(bf16);         // 10 MB
    float* probs = (float*)(ws + off); off += (size_t)NTOK * 8 * sizeof(float);
    float* wts   = (float*)(ws + off); off += (size_t)NTOK * 2 * sizeof(float);
    int* idx     = (int*)(ws + off);   off += (size_t)NTOK * 2 * sizeof(int);
    int* slot    = (int*)(ws + off);   off += (size_t)NK * sizeof(int);

    // 1) zero dispatched buffer (unfilled capacity slots contribute 0)
    moe_zero_u64<<<2048, 256, 0, stream>>>((unsigned long long*)disp,
                                           (long long)NSLOTS * DMODEL * 2 / 8);
    // 2) router
    moe_router<<<NTOK / 8, 256, 0, stream>>>(x, gate_w, probs, wts, idx);
    // 3) deterministic capacity scan
    moe_scan<<<1, 1024, 0, stream>>>(idx, slot);
    // 4) dispatch
    moe_dispatch<<<NK, 256, 0, stream>>>(x, slot, disp);
    // 5) expert FFNs (hidden buffer reused; stream order serializes)
    for (int e = 0; e < NEXP; ++e) {
        moe_gateup<<<dim3(FFN / 128, CAP / 128), 256, 0, stream>>>(
            disp + (size_t)e * CAP * DMODEL,
            w_gate + (size_t)e * DMODEL * FFN,
            w_up   + (size_t)e * DMODEL * FFN,
            hidden);
        moe_down<<<dim3(DMODEL / 128, CAP / 128), 256, 0, stream>>>(
            hidden,
            w_down + (size_t)e * FFN * DMODEL,
            eout + (size_t)e * CAP * DMODEL);
    }
    // 6) combine
    moe_combine<<<NTOK, 256, 0, stream>>>(eout, slot, wts, out);
    // 7) aux loss
    moe_aux<<<1, 256, 0, stream>>>(idx, probs, out + (size_t)NTOK * DMODEL);
}
